// VoxelContext3D_dsa_27848567947618
// MI455X (gfx1250) — matrix operand order repaired and checked by reading, not compiled, not run on hardware
//
#include <hip/hip_runtime.h>
#include <hip/hip_bf16.h>
#include <stdint.h>

#define B_    2
#define NVOX_ 16384
#define K_    2048
#define CIN_  64
#define G_    4
#define NX_   176
#define NY_   200

typedef __bf16 bf16;
typedef __attribute__((ext_vector_type(16))) __bf16 bf16x16;
typedef __attribute__((ext_vector_type(8)))  __bf16 bf16x8;
typedef __attribute__((ext_vector_type(8)))  float  f32x8;

__device__ __forceinline__ bf16 f2bf(float f) {
  union { float f; uint32_t u; } v; v.f = f;
  uint32_t r = v.u + 0x7FFFu + ((v.u >> 16) & 1u);
  union { uint16_t u; bf16 b; } o; o.u = (uint16_t)(r >> 16);
  return o.b;
}

__device__ __forceinline__ f32x8 wmma_bf16(bf16x16 a, bf16x16 b, f32x8 c) {
  return __builtin_amdgcn_wmma_f32_16x16x32_bf16(false, a, false, b, (short)0, c, false, false);
}

// A fragment: 16x32 bf16 tile from row-major source (lda elements, multiple of 8).
__device__ __forceinline__ bf16x16 load_a_frag(const bf16* A, int lda, int row0, int k0) {
  int lane = threadIdx.x & 31;
  int hf = lane >> 4;
  int r = lane & 15;
  const bf16* ap = A + (size_t)(row0 + r) * lda + k0 + hf * 16;
  bf16x8 lo = *(const bf16x8*)ap;
  bf16x8 hi = *(const bf16x8*)(ap + 8);
  return __builtin_shufflevector(lo, hi, 0, 1, 2, 3, 4, 5, 6, 7,
                                 8, 9, 10, 11, 12, 13, 14, 15);
}

// Packed B fragment: layout [kchunk][ntile][lane][16 bf16], 32B aligned per lane.
__device__ __forceinline__ bf16x16 load_bp_frag(const bf16* Wpk, int tilesN, int kc, int nt) {
  int lane = threadIdx.x & 31;
  return *(const bf16x16*)(Wpk + (((size_t)kc * tilesN + nt) * 32 + lane) * 16);
}

// B^T fragment: source row-major (N x Kc), Kc multiple of 32, used as (Kc x N).
__device__ __forceinline__ bf16x16 load_bt_frag(const bf16* Bs, int ld, int k0, int col0) {
  int lane = threadIdx.x & 31;
  int hf = lane >> 4;
  int c = lane & 15;
  return *(const bf16x16*)(Bs + (size_t)(col0 + c) * ld + k0 + hf * 16);
}

// ---------------- weight pack: f32 (Kin x N) -> fragment-packed zero-padded bf16 ----------------
__global__ void k_pack_w_frag(const float* src, bf16* dst, int Kin, int Kp, int N) {
  int t = blockIdx.x * blockDim.x + threadIdx.x;
  int tilesN = N >> 4;
  int tot = (Kp >> 5) * tilesN * 512;
  if (t >= tot) return;
  int e = t & 15;
  int lane = (t >> 4) & 31;
  int nt = (t >> 9) % tilesN;
  int kc = t / (512 * tilesN);
  int hf = lane >> 4, c = lane & 15;
  int k = (kc << 5) + hf * 16 + e;
  int col = nt * 16 + c;
  dst[t] = f2bf(k < Kin ? src[k * N + col] : 0.f);
}

// bf16 row-major (Kp x N) -> fragment-packed
__global__ void k_repack_frag(const bf16* src, bf16* dst, int Kp, int N) {
  int t = blockIdx.x * blockDim.x + threadIdx.x;
  int tilesN = N >> 4;
  int tot = (Kp >> 5) * tilesN * 512;
  if (t >= tot) return;
  int e = t & 15;
  int lane = (t >> 4) & 31;
  int nt = (t >> 9) % tilesN;
  int kc = t / (512 * tilesN);
  int hf = lane >> 4, c = lane & 15;
  int k = (kc << 5) + hf * 16 + e;
  dst[t] = src[(size_t)k * N + nt * 16 + c];
}

// ---------------- voxel centers ----------------
__global__ void k_xyz(const int* coords, float* xyz) {
  int t = blockIdx.x * blockDim.x + threadIdx.x;
  if (t >= B_ * NVOX_) return;
  const int* c = coords + (size_t)t * 4;
  xyz[t * 3 + 0] = ((float)c[3] + 0.5f) * 0.4f + 0.0f;
  xyz[t * 3 + 1] = ((float)c[2] + 0.5f) * 0.4f - 40.0f;
  xyz[t * 3 + 2] = ((float)c[1] + 0.5f) * 0.8f - 3.0f;
}

// ---------------- farthest point sampling ----------------
__global__ __launch_bounds__(1024) void k_fps(const float* xyz, int* kp_idx) {
  int b = blockIdx.x;
  const float* P = xyz + (size_t)b * NVOX_ * 3;
  int* out = kp_idx + b * K_;
  __shared__ float s_val[1024];
  __shared__ int s_idx[1024];
  __shared__ float s_last[3];
  int tid = threadIdx.x;
  float dreg[16];
#pragma unroll
  for (int j = 0; j < 16; ++j) dreg[j] = 1e10f;
  if (tid == 0) { out[0] = 0; s_last[0] = P[0]; s_last[1] = P[1]; s_last[2] = P[2]; }
  __syncthreads();
  for (int k = 1; k < K_; ++k) {
    float lx = s_last[0], ly = s_last[1], lz = s_last[2];
    float best = -1.f; int bi = 0;
#pragma unroll
    for (int j = 0; j < 16; ++j) {
      int i = tid + j * 1024;
      float dx = P[i * 3] - lx, dy = P[i * 3 + 1] - ly, dz = P[i * 3 + 2] - lz;
      float d = dx * dx + dy * dy + dz * dz;
      float dm = fminf(dreg[j], d);
      dreg[j] = dm;
      if (dm > best) { best = dm; bi = i; }
    }
    s_val[tid] = best; s_idx[tid] = bi;
    __syncthreads();
    for (int off = 512; off > 0; off >>= 1) {
      if (tid < off) {
        float ov = s_val[tid + off]; int oi = s_idx[tid + off];
        if (ov > s_val[tid] || (ov == s_val[tid] && oi < s_idx[tid])) { s_val[tid] = ov; s_idx[tid] = oi; }
      }
      __syncthreads();
    }
    if (tid == 0) {
      int n = s_idx[0]; out[k] = n;
      s_last[0] = P[n * 3]; s_last[1] = P[n * 3 + 1]; s_last[2] = P[n * 3 + 2];
    }
    __syncthreads();
  }
}

__global__ void k_gather_kps(const float* xyz, const int* kp_idx, float* kps) {
  int t = blockIdx.x * blockDim.x + threadIdx.x;
  if (t >= B_ * K_) return;
  int b = t / K_;
  int id = kp_idx[t];
  const float* p = xyz + ((size_t)b * NVOX_ + id) * 3;
  kps[t * 3] = p[0]; kps[t * 3 + 1] = p[1]; kps[t * 3 + 2] = p[2];
}

// ---------------- ball query: first ns hits in index order ----------------
__global__ void k_ballquery(const float* pts, int N, const float* q, int Q,
                            float r2, int ns, int* idx, int* anyv) {
  int wave = (blockIdx.x * blockDim.x + threadIdx.x) >> 5;
  int lane = threadIdx.x & 31;
  if (wave >= Q) return;
  float qx = q[wave * 3], qy = q[wave * 3 + 1], qz = q[wave * 3 + 2];
  int cnt = 0;
  int first = -1;
  for (int base = 0; base < N && cnt < ns; base += 32) {
    int p = base + lane;
    bool pred = false;
    if (p < N) {
      float dx = pts[p * 3] - qx, dy = pts[p * 3 + 1] - qy, dz = pts[p * 3 + 2] - qz;
      pred = (dx * dx + dy * dy + dz * dz) < r2;
    }
    unsigned mask = (unsigned)__ballot(pred);
    if (pred) {
      int slot = cnt + __popc(mask & ((1u << lane) - 1u));
      if (slot < ns) idx[(size_t)wave * ns + slot] = p;
    }
    if (first < 0 && mask) first = base + (__ffs(mask) - 1);
    cnt += __popc(mask);
  }
  if (cnt > ns) cnt = ns;
  if (lane == 0) anyv[wave] = (cnt > 0) ? 1 : 0;
  int fill = (first < 0) ? 0 : first;
  for (int s = cnt + lane; s < ns; s += 32) idx[(size_t)wave * ns + s] = fill;
}

// ---------------- gather group features -> bf16 (Q*ns, 96) ----------------
__global__ void k_gather_group(const float* pts, const float* feats, const float* q,
                               const int* idx, const int* anyv, bf16* g, int Q, int ns) {
  size_t t = (size_t)blockIdx.x * blockDim.x + threadIdx.x;
  size_t tot = (size_t)Q * ns * 96;
  if (t >= tot) return;
  int k = (int)(t % 96);
  size_t row = t / 96;
  int s = (int)(row % ns);
  size_t qi = row / ns;
  float v = 0.f;
  if (anyv[qi] && k < 67) {
    int id = idx[qi * ns + s];
    if (k < 3) v = pts[id * 3 + k] - q[qi * 3 + k];
    else       v = feats[(size_t)id * 64 + (k - 3)];
  }
  g[t] = f2bf(v);
}

// ---------------- WMMA GEMM: C = act(A @ W + b), W fragment-packed, N = TN*16 ----------------
template <int TN>
__global__ __launch_bounds__(256) void k_gemm(const bf16* A, const bf16* Wpk, const float* bias,
                                              float* Cf, bf16* Cb, int M, int Kp, int relu) {
  const int N = TN * 16;
  int wave = (blockIdx.x * blockDim.x + threadIdx.x) >> 5;
  int strips = M >> 4;
  if (wave >= strips) return;
  int tm = wave << 4;
  f32x8 acc[TN];
#pragma unroll
  for (int nt = 0; nt < TN; ++nt) acc[nt] = {};
  int nkc = Kp >> 5;
  for (int kc = 0; kc < nkc; ++kc) {
    bf16x16 va = load_a_frag(A, Kp, tm, kc << 5);
#pragma unroll
    for (int nt = 0; nt < TN; ++nt)
      acc[nt] = wmma_bf16(va, load_bp_frag(Wpk, TN, kc, nt), acc[nt]);
  }
  int lane = threadIdx.x & 31;
  int hf = lane >> 4, c = lane & 15;
#pragma unroll
  for (int nt = 0; nt < TN; ++nt) {
    float bv = bias ? bias[nt * 16 + c] : 0.f;
#pragma unroll
    for (int i = 0; i < 8; ++i) {
      float v = acc[nt][i] + bv;
      if (relu) v = fmaxf(v, 0.f);
      size_t o = (size_t)(tm + hf * 8 + i) * N + nt * 16 + c;
      if (Cf) Cf[o] = v;
      if (Cb) Cb[o] = f2bf(v);
    }
  }
}

// ---------------- WMMA GEMM NT: C(M,N) = A(M,Kc) @ Bsrc(N,Kc)^T, 16x64 per wave ----------------
__global__ __launch_bounds__(256) void k_gemm_nt(const bf16* A, const bf16* Bs, float* C,
                                                 int M, int N, int Kc) {
  int wave = (blockIdx.x * blockDim.x + threadIdx.x) >> 5;
  int bn = N >> 6;
  int blocks = (M >> 4) * bn;
  if (wave >= blocks) return;
  int tm = (wave / bn) << 4;
  int tn0 = (wave % bn) << 6;
  f32x8 acc[4];
#pragma unroll
  for (int nt = 0; nt < 4; ++nt) acc[nt] = {};
  for (int k0 = 0; k0 < Kc; k0 += 32) {
    bf16x16 va = load_a_frag(A, Kc, tm, k0);
#pragma unroll
    for (int nt = 0; nt < 4; ++nt)
      acc[nt] = wmma_bf16(va, load_bt_frag(Bs, Kc, k0, tn0 + nt * 16), acc[nt]);
  }
  int lane = threadIdx.x & 31;
  int hf = lane >> 4, c = lane & 15;
#pragma unroll
  for (int nt = 0; nt < 4; ++nt)
#pragma unroll
    for (int i = 0; i < 8; ++i)
      C[(size_t)(tm + hf * 8 + i) * N + tn0 + nt * 16 + c] = acc[nt][i];
}

// ---------------- max over ns samples ----------------
__global__ void k_maxns(const float* h, float* out, int Q, int ns, int C, int ostride, int ooff) {
  int t = blockIdx.x * blockDim.x + threadIdx.x;
  if (t >= Q * C) return;
  int q = t / C, c = t % C;
  const float* hp = h + ((size_t)q * ns) * C + c;
  float m = hp[0];
  for (int s = 1; s < ns; ++s) m = fmaxf(m, hp[(size_t)s * C]);
  out[(size_t)q * ostride + ooff + c] = m;
}

// ---------------- keypoint deform offsets ----------------
__global__ void k_offsets(const float* hd, const float* kps, const float* Woff,
                          const float* boff, float* dq) {
  int t = blockIdx.x * blockDim.x + threadIdx.x;
  if (t >= B_ * K_) return;
  const float* h = hd + (size_t)t * 32;
  float a0 = boff[0], a1 = boff[1], a2 = boff[2];
  for (int c = 0; c < 32; ++c) {
    float hv = h[c];
    a0 += hv * Woff[c * 3];
    a1 += hv * Woff[c * 3 + 1];
    a2 += hv * Woff[c * 3 + 2];
  }
  dq[t * 3 + 0] = kps[t * 3 + 0] + tanhf(a0) * 1.6f;
  dq[t * 3 + 1] = kps[t * 3 + 1] + tanhf(a1) * 1.6f;
  dq[t * 3 + 2] = kps[t * 3 + 2] + tanhf(a2) * 1.6f;
}

__global__ void k_tobf16(const float* x, bf16* y, int n) {
  int t = blockIdx.x * blockDim.x + threadIdx.x;
  if (t < n) y[t] = f2bf(x[t]);
}

// ---------------- softmax rows -> bf16 ----------------
__global__ void k_softmax(const float* S, bf16* attn, int N) {
  int row = blockIdx.x;
  const float* s = S + (size_t)row * N;
  __shared__ float red[256];
  int tid = threadIdx.x;
  float m = -1e30f;
  for (int j = tid; j < N; j += 256) m = fmaxf(m, s[j]);
  red[tid] = m; __syncthreads();
  for (int off = 128; off > 0; off >>= 1) {
    if (tid < off) red[tid] = fmaxf(red[tid], red[tid + off]);
    __syncthreads();
  }
  m = red[0]; __syncthreads();
  float sum = 0.f;
  for (int j = tid; j < N; j += 256) sum += __expf(s[j] - m);
  red[tid] = sum; __syncthreads();
  for (int off = 128; off > 0; off >>= 1) {
    if (tid < off) red[tid] += red[tid + off];
    __syncthreads();
  }
  float inv = 1.f / red[0];
  for (int j = tid; j < N; j += 256)
    attn[(size_t)row * N + j] = f2bf(__expf(s[j] - m) * inv);
}

// ---------------- grouped 1x1 conv (Wz) ----------------
__global__ void k_zconv(const float* o_rows, const float* Wz, const float* bz,
                        float* z, int layer) {
  int t = blockIdx.x * blockDim.x + threadIdx.x;
  if (t >= B_ * K_ * 64) return;
  int c = t & 63;
  int g = c >> 4, oc = c & 15;
  size_t bl = (size_t)(t >> 6);
  const float* ob = o_rows + bl * 64 + g * 16;
  const float* wz = Wz + (((size_t)layer * G_ + g) * 16) * 16 + oc;
  float acc = bz[layer * 64 + c];
  for (int cg = 0; cg < 16; ++cg) acc += ob[cg] * wz[cg * 16];
  z[t] = acc;
}

__global__ void k_gnstats(const float* z, float* stats) {
  int bg = blockIdx.x;
  int b = bg / G_, g = bg % G_;
  const float* zb = z + (size_t)b * K_ * 64;
  __shared__ float rs[256], rss[256];
  float s = 0.f, ss = 0.f;
  for (int t = threadIdx.x; t < K_ * 16; t += 256) {
    int l = t >> 4, cg = t & 15;
    float v = zb[(size_t)l * 64 + g * 16 + cg];
    s += v; ss += v * v;
  }
  rs[threadIdx.x] = s; rss[threadIdx.x] = ss;
  __syncthreads();
  for (int off = 128; off > 0; off >>= 1) {
    if (threadIdx.x < off) { rs[threadIdx.x] += rs[threadIdx.x + off]; rss[threadIdx.x] += rss[threadIdx.x + off]; }
    __syncthreads();
  }
  if (threadIdx.x == 0) {
    float n = (float)(K_ * 16);
    float mu = rs[0] / n;
    stats[bg * 2] = mu;
    stats[bg * 2 + 1] = rss[0] / n - mu * mu;
  }
}

__global__ void k_gnapply(const float* z, const float* stats, const float* gamma,
                          const float* beta, float* X, bf16* Xb, int layer) {
  int t = blockIdx.x * blockDim.x + threadIdx.x;
  if (t >= B_ * K_ * 64) return;
  int c = t & 63;
  int b = t / (K_ * 64);
  int g = c >> 4;
  float mu = stats[(b * G_ + g) * 2];
  float var = stats[(b * G_ + g) * 2 + 1];
  float zn = (z[t] - mu) * rsqrtf(var + 1e-5f);
  float xv = zn * gamma[layer * 64 + c] + beta[layer * 64 + c] + X[t];
  X[t] = xv;
  Xb[t] = f2bf(xv);
}

// ---------------- fused decode: gather + WMMA(96x64) + max over 16 ----------------
__device__ __forceinline__ float dec_feat(const float* kp, const float* ctx,
                                          float qx, float qy, float qz,
                                          int id, int av, int k) {
  if (!av || k >= 67) return 0.f;
  if (k < 3) {
    float qv = (k == 0) ? qx : ((k == 1) ? qy : qz);
    return kp[id * 3 + k] - qv;
  }
  return ctx[(size_t)id * 64 + (k - 3)];
}

__global__ __launch_bounds__(256) void k_decode(const float* kp, const float* ctx,
                                                const float* qs, const int* idx,
                                                const int* anyv, const bf16* Wdecb,
                                                const float* bdec, float* dec) {
  int wave = (blockIdx.x * blockDim.x + threadIdx.x) >> 5;
  if (wave >= NVOX_) return;
  int lane = threadIdx.x & 31;
  int hf = lane >> 4, r = lane & 15;
  int qi = wave;
  int av = anyv[qi];
  int id = idx[(size_t)qi * 16 + r];
  float qx = qs[qi * 3], qy = qs[qi * 3 + 1], qz = qs[qi * 3 + 2];
  bf16x16 afr[3];
#pragma unroll
  for (int ck = 0; ck < 3; ++ck) {
    int k0 = ck * 32 + hf * 16;
#pragma unroll
    for (int j = 0; j < 8; ++j) afr[ck][j] = f2bf(dec_feat(kp, ctx, qx, qy, qz, id, av, k0 + j));
#pragma unroll
    for (int j = 0; j < 8; ++j) afr[ck][8 + j] = f2bf(dec_feat(kp, ctx, qx, qy, qz, id, av, k0 + 8 + j));
  }
#pragma unroll
  for (int nt = 0; nt < 4; ++nt) {
    f32x8 acc = {};
#pragma unroll
    for (int ck = 0; ck < 3; ++ck)
      acc = wmma_bf16(afr[ck], load_bp_frag(Wdecb, 4, ck, nt), acc);
    float m = acc[0];
#pragma unroll
    for (int i = 1; i < 8; ++i) m = fmaxf(m, acc[i]);
    m = fmaxf(m, __shfl_xor(m, 16));
    int c = lane & 15;
    float v = fmaxf(m + bdec[nt * 16 + c], 0.f);
    if (lane < 16) dec[(size_t)qi * 64 + nt * 16 + c] = v;
  }
}

// ---------------- output: zero / scatter / avgpool ----------------
__global__ void k_zero(float* p, size_t n) {
  size_t t = (size_t)blockIdx.x * blockDim.x + threadIdx.x;
  if (t < n) p[t] = 0.f;
}

__global__ void k_scatter(const int* coords, const float* dec, float* out) {
  size_t t = (size_t)blockIdx.x * blockDim.x + threadIdx.x;
  if (t >= (size_t)B_ * NVOX_ * 64) return;
  int c = (int)(t & 63);
  size_t v = t >> 6;
  int b = (int)(v / NVOX_);
  int i = (int)(v % NVOX_);
  const int* cd = coords + ((size_t)b * NVOX_ + i) * 4;
  int lin = cd[1] + cd[2] * NX_ + cd[3];
  out[(size_t)b * 4505600 + (size_t)c * 70400 + lin] = dec[((size_t)b * NVOX_ + i) * 64 + c];
}

__global__ void k_avgpool(const float* bev, float* outHalf) {
  size_t t = (size_t)blockIdx.x * blockDim.x + threadIdx.x;
  size_t tot = (size_t)B_ * 128 * 100 * 88;
  if (t >= tot) return;
  int x = (int)(t % 88);
  int y = (int)((t / 88) % 100);
  int ch = (int)((t / 8800) % 128);
  int b = (int)(t / 1126400);
  size_t base = (size_t)b * 4505600 + (size_t)ch * 35200 + (size_t)(2 * y) * 176 + 2 * x;
  outHalf[t] = 0.25f * (bev[base] + bev[base + 1] + bev[base + 176] + bev[base + 177]);
}

static inline int cdiv(long a, long b) { return (int)((a + b - 1) / b); }

extern "C" void kernel_launch(void* const* d_in, const int* in_sizes, int n_in,
                              void* d_out, int out_size, void* d_ws, size_t ws_size,
                              hipStream_t stream) {
  const int* coords = (const int*)d_in[0];
  const float* vfeat = (const float*)d_in[1];
  const float* Wd0 = (const float*)d_in[2];  const float* bd0 = (const float*)d_in[3];
  const float* Wd1 = (const float*)d_in[4];  const float* bd1 = (const float*)d_in[5];
  const float* Woff = (const float*)d_in[6]; const float* boff = (const float*)d_in[7];
  const float* Wa00 = (const float*)d_in[8]; const float* ba00 = (const float*)d_in[9];
  const float* Wa01 = (const float*)d_in[10]; const float* ba01 = (const float*)d_in[11];
  const float* Wb00 = (const float*)d_in[12]; const float* bb00 = (const float*)d_in[13];
  const float* Wb01 = (const float*)d_in[14]; const float* bb01 = (const float*)d_in[15];
  const float* sa_Wt = (const float*)d_in[16];
  const float* sa_Wp = (const float*)d_in[17];
  const float* sa_Wg = (const float*)d_in[18];
  const float* sa_bt = (const float*)d_in[19];
  const float* sa_bp = (const float*)d_in[20];
  const float* sa_bg = (const float*)d_in[21];
  const float* sa_Wz = (const float*)d_in[22];
  const float* sa_bz = (const float*)d_in[23];
  const float* sa_gamma = (const float*)d_in[24];
  const float* sa_beta = (const float*)d_in[25];
  const float* Wdec = (const float*)d_in[26]; const float* bdec = (const float*)d_in[27];

  char* base = (char*)d_ws;
  size_t off = 0;
  auto alloc = [&](size_t bytes) -> void* {
    void* p = base + off;
    off = (off + bytes + 255) & ~(size_t)255;
    return p;
  };

  float* xyz   = (float*)alloc((size_t)B_ * NVOX_ * 3 * 4);
  int*   kpidx = (int*)alloc((size_t)B_ * K_ * 4);
  float* kps   = (float*)alloc((size_t)B_ * K_ * 3 * 4);
  float* dq    = (float*)alloc((size_t)B_ * K_ * 3 * 4);
  float* hd    = (float*)alloc((size_t)B_ * K_ * 32 * 4);
  int*   idx_s = (int*)alloc((size_t)B_ * K_ * 32 * 4);
  int*   anv_s = (int*)alloc((size_t)B_ * K_ * 4);
  int*   idx_d = (int*)alloc((size_t)B_ * NVOX_ * 16 * 4);
  int*   anv_d = (int*)alloc((size_t)B_ * NVOX_ * 4);
  bf16*  gbuf  = (bf16*)alloc((size_t)B_ * K_ * 32 * 96 * 2);
  bf16*  h1    = (bf16*)alloc((size_t)B_ * K_ * 32 * 32 * 2);
  float* h2    = (float*)alloc((size_t)B_ * K_ * 32 * 32 * 4);
  float* X     = (float*)alloc((size_t)B_ * K_ * 64 * 4);
  bf16*  Xb    = (bf16*)alloc((size_t)B_ * K_ * 64 * 2);
  bf16*  tb    = (bf16*)alloc((size_t)B_ * K_ * 64 * 2);
  bf16*  pb    = (bf16*)alloc((size_t)B_ * K_ * 64 * 2);
  bf16*  gvb   = (bf16*)alloc((size_t)B_ * K_ * 64 * 2);
  bf16*  gvp   = (bf16*)alloc((size_t)B_ * K_ * 64 * 2);
  float* S     = (float*)alloc((size_t)B_ * K_ * K_ * 4);
  bf16*  attn  = (bf16*)alloc((size_t)B_ * K_ * K_ * 2);
  float* orows = (float*)alloc((size_t)B_ * K_ * 64 * 4);
  float* zbuf  = (float*)alloc((size_t)B_ * K_ * 64 * 4);
  float* gnst  = (float*)alloc((size_t)B_ * G_ * 2 * 4);
  float* dec   = (float*)alloc((size_t)B_ * NVOX_ * 64 * 4);
  bf16*  Wd0b  = (bf16*)alloc(96 * 32 * 2);
  bf16*  Wd1b  = (bf16*)alloc(32 * 32 * 2);
  bf16*  Wa00b = (bf16*)alloc(96 * 32 * 2);
  bf16*  Wa01b = (bf16*)alloc(32 * 32 * 2);
  bf16*  Wb00b = (bf16*)alloc(96 * 32 * 2);
  bf16*  Wb01b = (bf16*)alloc(32 * 32 * 2);
  bf16*  Wtb   = (bf16*)alloc(3 * 64 * 64 * 2);
  bf16*  Wpb   = (bf16*)alloc(3 * 64 * 64 * 2);
  bf16*  Wgb   = (bf16*)alloc(3 * 64 * 64 * 2);
  bf16*  Wdecb = (bf16*)alloc(96 * 64 * 2);

  // ---- pack weights to fragment-major bf16 ----
  k_pack_w_frag<<<cdiv(96 * 32, 256), 256, 0, stream>>>(Wd0, Wd0b, 67, 96, 32);
  k_pack_w_frag<<<cdiv(32 * 32, 256), 256, 0, stream>>>(Wd1, Wd1b, 32, 32, 32);
  k_pack_w_frag<<<cdiv(96 * 32, 256), 256, 0, stream>>>(Wa00, Wa00b, 67, 96, 32);
  k_pack_w_frag<<<cdiv(32 * 32, 256), 256, 0, stream>>>(Wa01, Wa01b, 32, 32, 32);
  k_pack_w_frag<<<cdiv(96 * 32, 256), 256, 0, stream>>>(Wb00, Wb00b, 67, 96, 32);
  k_pack_w_frag<<<cdiv(32 * 32, 256), 256, 0, stream>>>(Wb01, Wb01b, 32, 32, 32);
  for (int i = 0; i < 3; ++i) {
    k_pack_w_frag<<<cdiv(64 * 64, 256), 256, 0, stream>>>(sa_Wt + (size_t)i * 4096, Wtb + (size_t)i * 4096, 64, 64, 64);
    k_pack_w_frag<<<cdiv(64 * 64, 256), 256, 0, stream>>>(sa_Wp + (size_t)i * 4096, Wpb + (size_t)i * 4096, 64, 64, 64);
    k_pack_w_frag<<<cdiv(64 * 64, 256), 256, 0, stream>>>(sa_Wg + (size_t)i * 4096, Wgb + (size_t)i * 4096, 64, 64, 64);
  }
  k_pack_w_frag<<<cdiv(96 * 64, 256), 256, 0, stream>>>(Wdec, Wdecb, 67, 96, 64);

  // ---- voxel centers + FPS + keypoints ----
  k_xyz<<<cdiv((long)B_ * NVOX_, 256), 256, 0, stream>>>(coords, xyz);
  k_fps<<<B_, 1024, 0, stream>>>(xyz, kpidx);
  k_gather_kps<<<cdiv((long)B_ * K_, 256), 256, 0, stream>>>(xyz, kpidx, kps);

  // ---- deformable group (r=1.6, ns=16) ----
  for (int b = 0; b < B_; ++b) {
    k_ballquery<<<cdiv((long)K_ * 32, 256), 256, 0, stream>>>(
        xyz + (size_t)b * NVOX_ * 3, NVOX_, kps + (size_t)b * K_ * 3, K_,
        2.56f, 16, idx_s + (size_t)b * K_ * 16, anv_s + (size_t)b * K_);
    k_gather_group<<<cdiv((long)K_ * 16 * 96, 256), 256, 0, stream>>>(
        xyz + (size_t)b * NVOX_ * 3, vfeat + (size_t)b * NVOX_ * 64,
        kps + (size_t)b * K_ * 3, idx_s + (size_t)b * K_ * 16, anv_s + (size_t)b * K_,
        gbuf + (size_t)b * K_ * 16 * 96, K_, 16);
  }
  {
    int M = B_ * K_ * 16;
    k_gemm<2><<<cdiv(M >> 4, 8), 256, 0, stream>>>(gbuf, Wd0b, bd0, nullptr, h1, M, 96, 1);
    k_gemm<2><<<cdiv(M >> 4, 8), 256, 0, stream>>>(h1, Wd1b, bd1, h2, nullptr, M, 32, 1);
    k_maxns<<<cdiv((long)B_ * K_ * 32, 256), 256, 0, stream>>>(h2, hd, B_ * K_, 16, 32, 32, 0);
  }
  k_offsets<<<cdiv((long)B_ * K_, 256), 256, 0, stream>>>(hd, kps, Woff, boff, dq);

  // ---- MSG group 0 (r=0.8, ns=16) -> X[:, 0:32] ----
  for (int b = 0; b < B_; ++b) {
    k_ballquery<<<cdiv((long)K_ * 32, 256), 256, 0, stream>>>(
        xyz + (size_t)b * NVOX_ * 3, NVOX_, dq + (size_t)b * K_ * 3, K_,
        0.64f, 16, idx_s + (size_t)b * K_ * 16, anv_s + (size_t)b * K_);
    k_gather_group<<<cdiv((long)K_ * 16 * 96, 256), 256, 0, stream>>>(
        xyz + (size_t)b * NVOX_ * 3, vfeat + (size_t)b * NVOX_ * 64,
        dq + (size_t)b * K_ * 3, idx_s + (size_t)b * K_ * 16, anv_s + (size_t)b * K_,
        gbuf + (size_t)b * K_ * 16 * 96, K_, 16);
  }
  {
    int M = B_ * K_ * 16;
    k_gemm<2><<<cdiv(M >> 4, 8), 256, 0, stream>>>(gbuf, Wa00b, ba00, nullptr, h1, M, 96, 1);
    k_gemm<2><<<cdiv(M >> 4, 8), 256, 0, stream>>>(h1, Wa01b, ba01, h2, nullptr, M, 32, 1);
    k_maxns<<<cdiv((long)B_ * K_ * 32, 256), 256, 0, stream>>>(h2, X, B_ * K_, 16, 32, 64, 0);
  }

  // ---- MSG group 1 (r=1.6, ns=32) -> X[:, 32:64] ----
  for (int b = 0; b < B_; ++b) {
    k_ballquery<<<cdiv((long)K_ * 32, 256), 256, 0, stream>>>(
        xyz + (size_t)b * NVOX_ * 3, NVOX_, dq + (size_t)b * K_ * 3, K_,
        2.56f, 32, idx_s + (size_t)b * K_ * 32, anv_s + (size_t)b * K_);
    k_gather_group<<<cdiv((long)K_ * 32 * 96, 256), 256, 0, stream>>>(
        xyz + (size_t)b * NVOX_ * 3, vfeat + (size_t)b * NVOX_ * 64,
        dq + (size_t)b * K_ * 3, idx_s + (size_t)b * K_ * 32, anv_s + (size_t)b * K_,
        gbuf + (size_t)b * K_ * 32 * 96, K_, 32);
  }
  {
    int M = B_ * K_ * 32;
    k_gemm<2><<<cdiv(M >> 4, 8), 256, 0, stream>>>(gbuf, Wb00b, bb00, nullptr, h1, M, 96, 1);
    k_gemm<2><<<cdiv(M >> 4, 8), 256, 0, stream>>>(h1, Wb01b, bb01, h2, nullptr, M, 32, 1);
    k_maxns<<<cdiv((long)B_ * K_ * 32, 256), 256, 0, stream>>>(h2, X, B_ * K_, 32, 32, 64, 32);
  }

  k_tobf16<<<cdiv((long)B_ * K_ * 64, 256), 256, 0, stream>>>(X, Xb, B_ * K_ * 64);

  // ---- 3 stacked SA blocks ----
  for (int i = 0; i < 3; ++i) {
    int Mtpg = B_ * K_;
    k_gemm<4><<<cdiv(Mtpg >> 4, 8), 256, 0, stream>>>(Xb, Wtb + (size_t)i * 4096, sa_bt + i * 64, nullptr, tb, Mtpg, 64, 0);
    k_gemm<4><<<cdiv(Mtpg >> 4, 8), 256, 0, stream>>>(Xb, Wpb + (size_t)i * 4096, sa_bp + i * 64, nullptr, pb, Mtpg, 64, 0);
    k_gemm<4><<<cdiv(Mtpg >> 4, 8), 256, 0, stream>>>(Xb, Wgb + (size_t)i * 4096, sa_bg + i * 64, nullptr, gvb, Mtpg, 64, 0);
    for (int b = 0; b < B_; ++b) {
      int blocksS = (K_ >> 4) * (K_ >> 6);
      k_gemm_nt<<<cdiv(blocksS, 8), 256, 0, stream>>>(
          tb + (size_t)b * K_ * 64, pb + (size_t)b * K_ * 64, S + (size_t)b * K_ * K_, K_, K_, 64);
      k_softmax<<<K_, 256, 0, stream>>>(S + (size_t)b * K_ * K_, attn + (size_t)b * K_ * K_, K_);
      k_repack_frag<<<cdiv((long)K_ * 64, 256), 256, 0, stream>>>(
          gvb + (size_t)b * K_ * 64, gvp + (size_t)b * K_ * 64, K_, 64);
      k_gemm<4><<<cdiv(K_ >> 4, 8), 256, 0, stream>>>(
          attn + (size_t)b * K_ * K_, gvp + (size_t)b * K_ * 64, nullptr,
          orows + (size_t)b * K_ * 64, nullptr, K_, K_, 0);
    }
    k_zconv<<<cdiv((long)B_ * K_ * 64, 256), 256, 0, stream>>>(orows, sa_Wz, sa_bz, zbuf, i);
    k_gnstats<<<B_ * G_, 256, 0, stream>>>(zbuf, gnst);
    k_gnapply<<<cdiv((long)B_ * K_ * 64, 256), 256, 0, stream>>>(zbuf, gnst, sa_gamma, sa_beta, X, Xb, i);
  }

  // ---- decode: ball query over keypoints + fused gather/WMMA/max ----
  for (int b = 0; b < B_; ++b) {
    k_ballquery<<<cdiv((long)NVOX_ * 32, 256), 256, 0, stream>>>(
        dq + (size_t)b * K_ * 3, K_, xyz + (size_t)b * NVOX_ * 3, NVOX_,
        2.56f, 16, idx_d + (size_t)b * NVOX_ * 16, anv_d + (size_t)b * NVOX_);
    k_decode<<<cdiv((long)NVOX_ * 32, 256), 256, 0, stream>>>(
        dq + (size_t)b * K_ * 3, X + (size_t)b * K_ * 64, xyz + (size_t)b * NVOX_ * 3,
        idx_d + (size_t)b * NVOX_ * 16, anv_d + (size_t)b * NVOX_, Wdecb, bdec,
        dec + (size_t)b * NVOX_ * 64);
  }

  // ---- BEV scatter + 2x2 avg pool ----
  float* outF = (float*)d_out;
  size_t bevN = (size_t)B_ * 4505600;
  k_zero<<<cdiv((long)bevN, 256), 256, 0, stream>>>(outF, bevN);
  k_scatter<<<cdiv((long)B_ * NVOX_ * 64, 256), 256, 0, stream>>>(coords, dec, outF);
  k_avgpool<<<cdiv((long)B_ * 128 * 100 * 88, 256), 256, 0, stream>>>(outF, outF + bevN);
}
